// Se3EquivariantTransformer_18708877541751
// MI455X (gfx1250) — matrix operand order repaired and checked by reading, not compiled, not run on hardware
//
#include <hip/hip_runtime.h>
#include <math.h>

#define N_NODES   50000
#define N_EDGES   400000
#define NFEAT     16
#define CHANNELS  64
#define HEADS     4
#define LAYERS    2
#define KDIM      16
#define SH_DIM    9
#define NUM_GRAPHS 16

typedef __attribute__((ext_vector_type(16))) _Float16 v16h;
typedef __attribute__((ext_vector_type(8)))  _Float16 v8h;
typedef __attribute__((ext_vector_type(8)))  float    v8f;

// ---------- monotone uint mapping for float atomic max ----------
__device__ __forceinline__ unsigned fmap(float f) {
    unsigned u = __float_as_uint(f);
    return (u & 0x80000000u) ? ~u : (u | 0x80000000u);
}
__device__ __forceinline__ float funmap(unsigned u) {
    unsigned b = (u & 0x80000000u) ? (u ^ 0x80000000u) : ~u;
    return __uint_as_float(b);
}

// ---------- zero output ----------
__global__ void zero_out_kernel(float* __restrict__ out, int n) {
    int i = blockIdx.x * blockDim.x + threadIdx.x;
    if (i < n) out[i] = 0.0f;
}

// ---------- spherical harmonics (l<=2, component norm) ----------
__global__ void sh_kernel(const float* __restrict__ rp, float* __restrict__ sh) {
    int e = blockIdx.x * blockDim.x + threadIdx.x;
    if (e >= N_EDGES) return;
    float px = rp[e * 3 + 0], py = rp[e * 3 + 1], pz = rp[e * 3 + 2];
    float r = sqrtf(px * px + py * py + pz * pz + 1e-12f);
    float x = px / r, y = py / r, z = pz / r;
    const float sq3 = 1.7320508075688772f;
    const float sq5 = 2.23606797749979f;
    const float sq15 = 3.872983346207417f;
    float* o = sh + (size_t)e * SH_DIM;
    o[0] = 1.0f;
    o[1] = sq3 * x; o[2] = sq3 * y; o[3] = sq3 * z;
    o[4] = sq15 * x * y;
    o[5] = sq15 * y * z;
    o[6] = (sq5 * 0.5f) * (3.0f * z * z - 1.0f);
    o[7] = sq15 * x * z;
    o[8] = (sq15 * 0.5f) * (x * x - y * y);
}

// ---------- embed: h0 = nf @ W_embed + b ----------
__global__ void embed_kernel(const float* __restrict__ nf, const float* __restrict__ W,
                             const float* __restrict__ b, float* __restrict__ h0) {
    int i = blockIdx.x * blockDim.x + threadIdx.x;   // N*64
    int n = i >> 6, c = i & 63;
    float acc = b[c];
#pragma unroll
    for (int j = 0; j < NFEAT; ++j) acc += nf[n * NFEAT + j] * W[j * CHANNELS + c];
    h0[i] = acc;
}

// ---------- per-head reset: h = h0 ; h_half = f16(h0) ----------
__global__ void copy_head_kernel(const float* __restrict__ h0, float* __restrict__ h,
                                 _Float16* __restrict__ hh) {
    int i = blockIdx.x * blockDim.x + threadIdx.x;   // N*64
    float v = h0[i];
    h[i] = v;
    hh[i] = (_Float16)v;
}

// ---------- per-iteration weight prep: transpose + f32->f16 ----------
// wqT/wkT: [16 cols][64 K] halves ; wvT: [64 cols][64 K] halves
__global__ void prep_weights_kernel(const float* __restrict__ wq, const float* __restrict__ wk,
                                    const float* __restrict__ wv,
                                    _Float16* __restrict__ wqT, _Float16* __restrict__ wkT,
                                    _Float16* __restrict__ wvT) {
    int i = blockIdx.x * blockDim.x + threadIdx.x;   // 1024 + 4096
    if (i < KDIM * CHANNELS) {
        int c = i >> 6, kk = i & 63;
        wqT[c * 64 + kk] = (_Float16)wq[kk * KDIM + c];
        wkT[c * 64 + kk] = (_Float16)wk[kk * KDIM + c];
    } else {
        int j = i - KDIM * CHANNELS;
        if (j < CHANNELS * CHANNELS) {
            int c = j >> 6, kk = j & 63;
            wvT[c * 64 + kk] = (_Float16)wv[kk * CHANNELS + c];
        }
    }
}

// ---------- WMMA B-fragment from transposed f16 weights ----------
// lane's 16 elements = K 8h..8h+7 and 16+8h..16+8h+7 of one column -> two 16B loads
__device__ __forceinline__ v16h load_bfragT(const _Float16* __restrict__ Wt,
                                            int colTile, int ktile, int lane) {
    int col = (lane & 15) + colTile * 16;
    int kbase = ktile * 32 + (lane >> 4) * 8;        // lanes 16-31 hold K 8-15 & 24-31
    const v8h* p = (const v8h*)(Wt + col * 64 + kbase);
    v8h lo = p[0], hi = p[2];
    v16h b;
#pragma unroll
    for (int i = 0; i < 8; ++i) { b[i] = lo[i]; b[8 + i] = hi[i]; }
    return b;
}

// ---------- q,k,v GEMMs via v_wmma_f32_16x16x32_f16 ----------
// one wave = 16 node rows; compute+store q, then k, then each v tile
// sequentially to keep only one B fragment + one accumulator live.
__global__ void qkv_wmma_kernel(const _Float16* __restrict__ hh,
                                const _Float16* __restrict__ wqT,
                                const _Float16* __restrict__ wkT,
                                const _Float16* __restrict__ wvT,
                                float* __restrict__ q, float* __restrict__ k,
                                float* __restrict__ v, int ntiles) {
    int wave = (blockIdx.x * blockDim.x + threadIdx.x) >> 5;
    int lane = threadIdx.x & 31;
    if (wave >= ntiles) return;                      // wave-uniform: EXEC all-ones inside
    int m = lane & 15, hi = lane >> 4;
    size_t r0 = (size_t)wave * 16;

    // A fragments (16x32 f16, ISA layout): lane<16 -> K 0-7 & 16-23 ; lane>=16 -> K 8-15 & 24-31
    v16h a0, a1;
    {
        const _Float16* base = hh + (r0 + m) * CHANNELS + 8 * hi;
        v8h x0 = *(const v8h*)(base + 0);
        v8h x1 = *(const v8h*)(base + 16);
        v8h x2 = *(const v8h*)(base + 32);
        v8h x3 = *(const v8h*)(base + 48);
#pragma unroll
        for (int i = 0; i < 8; ++i) {
            a0[i] = x0[i]; a0[8 + i] = x1[i];
            a1[i] = x2[i]; a1[8 + i] = x3[i];
        }
    }

    int col = lane & 15;

    // q = h @ Wq  (K=64 -> 2 k-steps)
    {
        v8f c = {};
        c = __builtin_amdgcn_wmma_f32_16x16x32_f16(false, a0, false,
                load_bfragT(wqT, 0, 0, lane), (short)0, c, false, false);
        c = __builtin_amdgcn_wmma_f32_16x16x32_f16(false, a1, false,
                load_bfragT(wqT, 0, 1, lane), (short)0, c, false, false);
#pragma unroll
        for (int j = 0; j < 8; ++j) q[(r0 + j + 8 * hi) * KDIM + col] = c[j];
    }
    // k = h @ Wk
    {
        v8f c = {};
        c = __builtin_amdgcn_wmma_f32_16x16x32_f16(false, a0, false,
                load_bfragT(wkT, 0, 0, lane), (short)0, c, false, false);
        c = __builtin_amdgcn_wmma_f32_16x16x32_f16(false, a1, false,
                load_bfragT(wkT, 0, 1, lane), (short)0, c, false, false);
#pragma unroll
        for (int j = 0; j < 8; ++j) k[(r0 + j + 8 * hi) * KDIM + col] = c[j];
    }
    // v = h @ Wv  (4 column tiles x 2 k-steps)
#pragma unroll
    for (int t = 0; t < 4; ++t) {
        v8f c = {};
        c = __builtin_amdgcn_wmma_f32_16x16x32_f16(false, a0, false,
                load_bfragT(wvT, t, 0, lane), (short)0, c, false, false);
        c = __builtin_amdgcn_wmma_f32_16x16x32_f16(false, a1, false,
                load_bfragT(wvT, t, 1, lane), (short)0, c, false, false);
#pragma unroll
        for (int j = 0; j < 8; ++j)
            v[(r0 + j + 8 * hi) * CHANNELS + t * 16 + col] = c[j];
    }
}

// ---------- per-iteration reset: num=0, den=0, m=-inf(mapped 0) ----------
__global__ void init_iter_kernel(float* __restrict__ num, float* __restrict__ den,
                                 unsigned* __restrict__ mu) {
    int i = blockIdx.x * blockDim.x + threadIdx.x;   // N*64
    num[i] = 0.0f;
    if ((i & 63) == 0) {
        int n = i >> 6;
        den[n] = 0.0f;
        mu[n] = 0u;                                   // maps below every float
    }
}

// ---------- edge pass 1: logits + segment max ----------
__global__ void edge_logits_kernel(const int* __restrict__ src, const int* __restrict__ dst,
                                   const float* __restrict__ q, const float* __restrict__ k,
                                   const float* __restrict__ sh, const float* __restrict__ wsl,
                                   float* __restrict__ lg, unsigned* __restrict__ mu) {
    int e = blockIdx.x * blockDim.x + threadIdx.x;
    if (e >= N_EDGES) return;
    int s = src[e], d = dst[e];
    const float* qd = q + (size_t)d * KDIM;
    const float* ks = k + (size_t)s * KDIM;
    float acc = 0.0f;
#pragma unroll
    for (int i = 0; i < KDIM; ++i) acc += qd[i] * ks[i];
    acc *= 0.25f;                                    // 1/sqrt(KDIM)
    const float* she = sh + (size_t)e * SH_DIM;
#pragma unroll
    for (int j = 0; j < SH_DIM; ++j) acc += she[j] * wsl[j];
    lg[e] = acc;
    atomicMax(&mu[d], fmap(acc));
}

// ---------- edge pass 2: exp + SH-modulated message scatter ----------
// block = 256 threads = 4 edges x 64 channels
__global__ void edge_scatter_kernel(const int* __restrict__ src, const int* __restrict__ dst,
                                    const float* __restrict__ v, const float* __restrict__ sh,
                                    const float* __restrict__ lg, const unsigned* __restrict__ mu,
                                    const float* __restrict__ wsv,
                                    float* __restrict__ num, float* __restrict__ den) {
    __shared__ float sW[SH_DIM * CHANNELS];
    for (int i = threadIdx.x; i < SH_DIM * CHANNELS; i += blockDim.x) sW[i] = wsv[i];
    __syncthreads();
    int e = blockIdx.x * 4 + (threadIdx.x >> 6);
    int c = threadIdx.x & 63;
    int s = src[e], d = dst[e];
    float m = funmap(mu[d]);                          // every dst touched has >=1 edge
    float a = __expf(lg[e] - m);
    const float* she = sh + (size_t)e * SH_DIM;
    float sv = 0.0f;
#pragma unroll
    for (int j = 0; j < SH_DIM; ++j) sv += she[j] * sW[j * CHANNELS + c];
    atomicAdd(&num[(size_t)d * CHANNELS + c], a * v[(size_t)s * CHANNELS + c] * sv);
    if (c == 0) atomicAdd(&den[d], a);
}

// ---------- node update: h += num/(den+eps); refresh f16 mirror ----------
__global__ void update_kernel(float* __restrict__ h, _Float16* __restrict__ hh,
                              const float* __restrict__ num, const float* __restrict__ den) {
    int i = blockIdx.x * blockDim.x + threadIdx.x;   // N*64
    int n = i >> 6;
    float val = h[i] + num[i] / (den[n] + 1e-9f);
    h[i] = val;
    hh[i] = (_Float16)val;
}

// ---------- pooled output per head ----------
__global__ void pool_kernel(const float* __restrict__ h, const int* __restrict__ bid,
                            float* __restrict__ out, int hd) {
    int i = blockIdx.x * blockDim.x + threadIdx.x;   // N*64
    int n = i >> 6, c = i & 63;
    atomicAdd(&out[bid[n] * (HEADS * CHANNELS) + hd * CHANNELS + c], h[i]);
}

extern "C" void kernel_launch(void* const* d_in, const int* in_sizes, int n_in,
                              void* d_out, int out_size, void* d_ws, size_t ws_size,
                              hipStream_t stream) {
    const float* nf   = (const float*)d_in[0];
    const float* rp   = (const float*)d_in[1];
    const int*   ei   = (const int*)d_in[2];
    const int*   bid  = (const int*)d_in[3];
    const float* Wemb = (const float*)d_in[4];
    const float* bemb = (const float*)d_in[5];
    const float* Wq   = (const float*)d_in[6];
    const float* Wk   = (const float*)d_in[7];
    const float* Wv   = (const float*)d_in[8];
    const float* Wsl  = (const float*)d_in[9];
    const float* Wsv  = (const float*)d_in[10];
    float* out = (float*)d_out;

    const int* src = ei;
    const int* dst = ei + N_EDGES;

    // workspace carve-up (all offsets multiples of 8 floats -> 32B aligned)
    float* ws = (float*)d_ws;
    size_t off = 0;
    float*    h0  = ws + off; off += (size_t)N_NODES * CHANNELS;
    float*    h   = ws + off; off += (size_t)N_NODES * CHANNELS;
    float*    q   = ws + off; off += (size_t)N_NODES * KDIM;
    float*    k   = ws + off; off += (size_t)N_NODES * KDIM;
    float*    v   = ws + off; off += (size_t)N_NODES * CHANNELS;
    float*    sh  = ws + off; off += (size_t)N_EDGES * SH_DIM;
    float*    lg  = ws + off; off += (size_t)N_EDGES;
    float*    num = ws + off; off += (size_t)N_NODES * CHANNELS;
    float*    den = ws + off; off += (size_t)N_NODES;
    unsigned* mu  = (unsigned*)(ws + off); off += (size_t)N_NODES;
    _Float16* hh  = (_Float16*)(ws + off); off += (size_t)N_NODES * CHANNELS / 2;
    _Float16* wqT = (_Float16*)(ws + off); off += (size_t)KDIM * CHANNELS / 2;
    _Float16* wkT = (_Float16*)(ws + off); off += (size_t)KDIM * CHANNELS / 2;
    _Float16* wvT = (_Float16*)(ws + off); off += (size_t)CHANNELS * CHANNELS / 2;

    const int NC_BLOCKS = (N_NODES * CHANNELS) / 256;           // 12500
    const int NTILES    = N_NODES / 16;                         // 3125
    const int QKV_BLK   = (NTILES + 7) / 8;                     // 8 waves/block
    const int E_BLOCKS  = (N_EDGES + 255) / 256;
    const int ES_BLOCKS = N_EDGES / 4;                          // 4 edges/block
    const int PW_BLOCKS = (KDIM * CHANNELS + CHANNELS * CHANNELS + 255) / 256;

    zero_out_kernel<<<(NUM_GRAPHS * HEADS * CHANNELS + 255) / 256, 256, 0, stream>>>(
        out, NUM_GRAPHS * HEADS * CHANNELS);
    sh_kernel<<<E_BLOCKS, 256, 0, stream>>>(rp, sh);
    embed_kernel<<<NC_BLOCKS, 256, 0, stream>>>(nf, Wemb, bemb, h0);

    for (int hd = 0; hd < HEADS; ++hd) {
        copy_head_kernel<<<NC_BLOCKS, 256, 0, stream>>>(h0, h, hh);
        for (int l = 0; l < LAYERS; ++l) {
            int il = hd * LAYERS + l;
            const float* wq  = Wq  + (size_t)il * CHANNELS * KDIM;
            const float* wk  = Wk  + (size_t)il * CHANNELS * KDIM;
            const float* wv  = Wv  + (size_t)il * CHANNELS * CHANNELS;
            const float* wsl = Wsl + (size_t)il * SH_DIM;
            const float* wsv = Wsv + (size_t)il * SH_DIM * CHANNELS;

            prep_weights_kernel<<<PW_BLOCKS, 256, 0, stream>>>(wq, wk, wv, wqT, wkT, wvT);
            qkv_wmma_kernel<<<QKV_BLK, 256, 0, stream>>>(hh, wqT, wkT, wvT, q, k, v, NTILES);
            init_iter_kernel<<<NC_BLOCKS, 256, 0, stream>>>(num, den, mu);
            edge_logits_kernel<<<E_BLOCKS, 256, 0, stream>>>(src, dst, q, k, sh, wsl, lg, mu);
            edge_scatter_kernel<<<ES_BLOCKS, 256, 0, stream>>>(src, dst, v, sh, lg, mu, wsv,
                                                               num, den);
            update_kernel<<<NC_BLOCKS, 256, 0, stream>>>(h, hh, num, den);
        }
        pool_kernel<<<NC_BLOCKS, 256, 0, stream>>>(h, bid, out, hd);
    }
}
